// RHAN_29231547417247
// MI455X (gfx1250) — compile-verified, matrix operand order checked by reading, not run on hardware
//
#include <hip/hip_runtime.h>
#include <hip/hip_bf16.h>

// ---------------------------------------------------------------------------
// HAN GNN forward for MI455X (gfx1250, wave32).
//  - Dense projections / k_lin run through v_wmma_f32_16x16x32_bf16
//    (bf16 A/B, f32 accumulate), 16x16 tile per wave, K=128 in 4 WMMA steps.
//  - A-tile (16x128, shared by all 8 waves of a block) is staged into LDS via
//    GLOBAL_LOAD_ASYNC_TO_LDS_B128 (ASYNCcnt path) -> 8x less global traffic.
//  - Edge softmax-aggregation: wave-per-edge float4 gather + native
//    global_atomic_add_f32 scatter; segment-max via ordered-uint atomicMax.
//  - All wave reductions use width-32 shuffles (wave32 only on CDNA5).
// ---------------------------------------------------------------------------

#define NN   50000      // nodes per type
#define CH   128        // feature dim (F_IN == H1 == H2)
#define NEDG 500000     // edges per edge type
#define NLBL 200000     // label edges per edge type

typedef __bf16 bf16_t;
typedef bf16_t v16bf __attribute__((ext_vector_type(16)));
typedef bf16_t v8bf  __attribute__((ext_vector_type(8)));
typedef float  v8f   __attribute__((ext_vector_type(8)));

__device__ __forceinline__ unsigned f2ord(float f) {
  unsigned u = __float_as_uint(f);
  return (u & 0x80000000u) ? ~u : (u | 0x80000000u);
}
__device__ __forceinline__ float ord2f(unsigned o) {
  return __uint_as_float((o & 0x80000000u) ? (o & 0x7fffffffu) : ~o);
}
#define ORD_NEG_INF 0x007FFFFFu   // f2ord(-inf)

// ---------------------------------------------------------------- utilities
__global__ void k_fill_u32(unsigned* __restrict__ p, unsigned v, int n) {
  int t = blockIdx.x * 256 + threadIdx.x;
  if (t < n) p[t] = v;
}

__global__ void k_f32_to_bf16(const float* __restrict__ x, bf16_t* __restrict__ y, int n) {
  int t = blockIdx.x * 256 + threadIdx.x;
  if (t < n) y[t] = (bf16_t)x[t];
}

// Wt[n*128 + k] = bf16(W[k*128 + n])  (transpose so B columns are contiguous)
__global__ void k_wtrans(const float* __restrict__ w, bf16_t* __restrict__ wt) {
  int t = blockIdx.x * 256 + threadIdx.x;
  if (t < CH * CH) {
    int k = t / CH, n = t % CH;
    wt[n * CH + k] = (bf16_t)w[t];
  }
}

// ------------------------------------------------------------- WMMA GEMM
// Y(3125*16 x 128) = act( X(bf16, row-major) @ W + bias )
// Block: 256 threads = 8 waves; wave w owns column tile n0 = 16*w.
// A tile (16 rows x 128 k, 4 KB) is async-DMA'd to LDS once per block.
// A frag (16x32 bf16): lane l holds row m0+(l&15), K = kb..kb+7 and kb+16..kb+23
//   with kb = k0 + 8*(l>>4)  (documented 16-bit A layout).
// B frag: lane l holds column n0+(l&15) of W (from transposed Wt), same K split.
// C/D f32: vgpr r, lane l -> element (r + 8*(l>>4), l&15).
__global__ void k_wmma_gemm(const bf16_t* __restrict__ X, const bf16_t* __restrict__ Wt,
                            const float* __restrict__ bias, float* __restrict__ Y,
                            int act /*0 none, 2 tanh*/) {
  __shared__ bf16_t As[16 * CH];            // 4 KB staged A tile
  const int tid  = threadIdx.x;
  const int lane = tid & 31;
  const int wave = tid >> 5;
  const int m0   = blockIdx.x << 4;
  const int n0   = wave << 4;
  const int hlf  = lane >> 4;
  const int l16  = lane & 15;
  const int bcol = n0 + l16;

  // ---- async-stage the 16x128 A tile: one b128 per thread (256*16B = 4 KB)
  {
    const bf16_t* gsrc = X + (size_t)(m0 + (tid >> 4)) * CH + ((tid & 15) << 3);
    unsigned ldsa = (unsigned)(uintptr_t)As + (unsigned)tid * 16u;  // 32-bit LDS byte address
    asm volatile("global_load_async_to_lds_b128 %0, %1, off"
                 :: "v"(ldsa), "v"(gsrc) : "memory");
    asm volatile("s_wait_asynccnt 0x0" ::: "memory");
  }
  __syncthreads();

  v8f acc;
  const float bv = bias[n0 + l16];
#pragma unroll
  for (int r = 0; r < 8; ++r) acc[r] = bv;

#pragma unroll
  for (int k0 = 0; k0 < CH; k0 += 32) {
    const int kb = k0 + hlf * 8;
    v8bf alo = *(const v8bf*)(As + l16 * CH + kb);          // ds_load_b128
    v8bf ahi = *(const v8bf*)(As + l16 * CH + kb + 16);
    v8bf blo = *(const v8bf*)(Wt + (size_t)bcol * CH + kb); // global_load_b128
    v8bf bhi = *(const v8bf*)(Wt + (size_t)bcol * CH + kb + 16);
    v16bf a, b;
#pragma unroll
    for (int i = 0; i < 8; ++i) { a[i] = alo[i]; a[i + 8] = ahi[i];
                                  b[i] = blo[i]; b[i + 8] = bhi[i]; }
    acc = __builtin_amdgcn_wmma_f32_16x16x32_bf16(false, a, false, b,
                                                  (short)0, acc, false, false);
  }

#pragma unroll
  for (int r = 0; r < 8; ++r) {
    float v = acc[r];
    if (act == 2) v = tanhf(v);
    Y[(size_t)(m0 + r + hlf * 8) * CH + n0 + l16] = v;
  }
}

// -------------------------------------------------- per-node attention dot
// out[n] = dot(H[n,:], att)   one wave per node, float4 per lane
__global__ void k_attdot(const float* __restrict__ H, const float* __restrict__ att,
                         float* __restrict__ out, int n) {
  int node = blockIdx.x * 8 + (threadIdx.x >> 5);
  if (node >= n) return;
  int lane = threadIdx.x & 31;
  float4 h = *(const float4*)(H + (size_t)node * CH + lane * 4);
  float4 a = *(const float4*)(att + lane * 4);
  float s = h.x * a.x + h.y * a.y + h.z * a.z + h.w * a.w;
#pragma unroll
  for (int m = 16; m; m >>= 1) s += __shfl_xor(s, m, 32);
  if (lane == 0) out[node] = s;
}

// ------------------------------------------------------------ edge passes
__global__ void k_edge_max(const long long* __restrict__ ei, const float* __restrict__ as,
                           const float* __restrict__ ad, float* __restrict__ alpha,
                           unsigned* __restrict__ amax, int ne) {
  int e = blockIdx.x * 256 + threadIdx.x;
  if (e >= ne) return;
  long long r = ei[e], c = ei[ne + e];
  float al = as[r] + ad[c];
  al = al > 0.f ? al : 0.2f * al;              // leaky_relu(0.2)
  alpha[e] = al;
  atomicMax(&amax[c], f2ord(al));
}

__global__ void k_edge_exp(const long long* __restrict__ ei, const float* __restrict__ alpha,
                           const unsigned* __restrict__ amax, float* __restrict__ eexp,
                           float* __restrict__ denom, int ne) {
  int e = blockIdx.x * 256 + threadIdx.x;
  if (e >= ne) return;
  long long c = ei[ne + e];
  float ev = __expf(alpha[e] - ord2f(amax[c]));
  eexp[e] = ev;
  atomicAdd(&denom[c], ev);
}

__global__ void k_edge_msg(const long long* __restrict__ ei, const float* __restrict__ eexp,
                           const float* __restrict__ denom, const float* __restrict__ srcH,
                           float* __restrict__ out, int ne) {
  int e = blockIdx.x * 8 + (threadIdx.x >> 5);
  if (e >= ne) return;
  int lane = threadIdx.x & 31;
  long long r = ei[e], c = ei[ne + e];
  const float* xp = srcH + (size_t)r * CH + lane * 4;
  __builtin_prefetch(xp, 0, 3);                 // global_prefetch_b8
  float w = eexp[e] / fmaxf(denom[c], 1e-16f);
  float4 x = *(const float4*)xp;
  float* o = out + (size_t)c * CH + lane * 4;
  atomicAdd(o + 0, x.x * w);
  atomicAdd(o + 1, x.y * w);
  atomicAdd(o + 2, x.z * w);
  atomicAdd(o + 3, x.w * w);
}

__global__ void k_relu(float* __restrict__ p, int n) {
  int t = blockIdx.x * 256 + threadIdx.x;
  if (t < n) p[t] = fmaxf(p[t], 0.f);
}

// ------------------------------------------- semantic attention reductions
// accumulate column sums of x (NN x 128) into acc[0..127]; 64 rows per block
__global__ void k_colsum(const float* __restrict__ x, float* __restrict__ acc) {
  __shared__ float sm[256];
  int rbase = blockIdx.x * 64;
  int c  = threadIdx.x & 127;
  int rh = threadIdx.x >> 7;
  int rend = rbase + 64; if (rend > NN) rend = NN;
  float s = 0.f;
  for (int r = rbase + rh; r < rend; r += 2) s += x[(size_t)r * CH + c];
  sm[threadIdx.x] = s;
  __syncthreads();
  if (threadIdx.x < 128) atomicAdd(&acc[c], sm[threadIdx.x] + sm[threadIdx.x + 128]);
}

// attn = softmax_k( (1/NN) * sum_c q[c]*kmean[k,c] ); one wave
__global__ void k_attn(const float* __restrict__ q, const float* __restrict__ kmean,
                       float* __restrict__ attn, int K) {
  int lane = threadIdx.x;
  float s[2] = {0.f, 0.f};
  for (int k = 0; k < K; ++k) {
    float p = 0.f;
    for (int c = lane; c < CH; c += 32) p += q[c] * kmean[k * CH + c];
#pragma unroll
    for (int m = 16; m; m >>= 1) p += __shfl_xor(p, m, 32);
    s[k] = p * (1.0f / NN);
  }
  if (lane == 0) {
    float mx = s[0];
    for (int k = 1; k < K; ++k) mx = fmaxf(mx, s[k]);
    float e[2], tot = 0.f;
    for (int k = 0; k < K; ++k) { e[k] = __expf(s[k] - mx); tot += e[k]; }
    for (int k = 0; k < K; ++k) attn[k] = e[k] / tot;
  }
}

// res = relu( attn0*o0 (+ attn1*o1) )
__global__ void k_combine(const float* __restrict__ o0, const float* __restrict__ o1,
                          const float* __restrict__ attn, int K,
                          float* __restrict__ res, int n) {
  int t = blockIdx.x * 256 + threadIdx.x;
  if (t >= n) return;
  float v = attn[0] * o0[t];
  if (K > 1) v += attn[1] * o1[t];
  res[t] = fmaxf(v, 0.f);
}

// ------------------------------------------------------------- post + score
__global__ void k_post(const float* __restrict__ H, const float* __restrict__ w,
                       const float* __restrict__ b, float* __restrict__ out, int n) {
  int node = blockIdx.x * 8 + (threadIdx.x >> 5);
  if (node >= n) return;
  int lane = threadIdx.x & 31;
  float4 h = *(const float4*)(H + (size_t)node * CH + lane * 4);
  float hv[4] = {h.x, h.y, h.z, h.w};
  float s0 = 0.f, s1 = 0.f;
#pragma unroll
  for (int j = 0; j < 4; ++j) {
    s0 += hv[j] * w[(lane * 4 + j) * 2 + 0];
    s1 += hv[j] * w[(lane * 4 + j) * 2 + 1];
  }
#pragma unroll
  for (int m = 16; m; m >>= 1) { s0 += __shfl_xor(s0, m, 32); s1 += __shfl_xor(s1, m, 32); }
  if (lane == 0) { out[node * 2] = s0 + b[0]; out[node * 2 + 1] = s1 + b[1]; }
}

__global__ void k_score(const long long* __restrict__ lbl, const float* __restrict__ headP,
                        const float* __restrict__ tailP, const float* __restrict__ rel,
                        int ridx, float* __restrict__ out, int n) {
  int e = blockIdx.x * 256 + threadIdx.x;
  if (e >= n) return;
  long long s = lbl[e], d = lbl[n + e];
  float hr = headP[s * 2], hi = headP[s * 2 + 1];
  float tr = tailP[d * 2], ti = tailP[d * 2 + 1];
  float rr = rel[ridx * 2], ri = rel[ridx * 2 + 1];
  out[e] = hr * rr * tr + hi * rr * ti + hr * ri * ti - hi * ri * tr;
}

// ---------------------------------------------------------------------------
extern "C" void kernel_launch(void* const* d_in, const int* in_sizes, int n_in,
                              void* d_out, int out_size, void* d_ws, size_t ws_size,
                              hipStream_t stream) {
  (void)in_sizes; (void)n_in; (void)out_size; (void)ws_size;

  const size_t NC = (size_t)NN * CH;
  const int iNC = (int)NC;

  // ---- inputs (setup_inputs dict order; params pytree flattened w/ sorted keys)
  const float*     x_user = (const float*)d_in[0];
  const float*     x_item = (const float*)d_in[1];
  const long long* ei[3]  = {(const long long*)d_in[2], (const long long*)d_in[3],
                             (const long long*)d_in[4]};
  const long long* lbl[3] = {(const long long*)d_in[5], (const long long*)d_in[6],
                             (const long long*)d_in[7]};
  // conv leaves (sorted): att_dst{irbu,ufu,uri}, att_src{irbu,ufu,uri},
  //                       k_lin{b,w}, proj{item{b,w}, user{b,w}}, q
  const int C1 = 8, C2 = 21;
  const float* post_b  = (const float*)d_in[34];
  const float* post_w  = (const float*)d_in[35];
  const float* rel_emb = (const float*)d_in[36];

  // ---- workspace layout
  char* Wp = (char*)d_ws;
  size_t off = 0;
  auto A = [&](size_t bytes) -> char* {
    char* p = Wp + off;
    off += (bytes + 255) & ~(size_t)255;
    return p;
  };
  bf16_t* xb_u  = (bf16_t*)A(NC * 2);
  bf16_t* xb_i  = (bf16_t*)A(NC * 2);
  bf16_t* wb    = (bf16_t*)A((size_t)CH * CH * 2);
  float*  hp_u  = (float*)A(NC * 4);
  float*  hp_i  = (float*)A(NC * 4);
  float*  outb[3] = {(float*)A(NC * 4), (float*)A(NC * 4), (float*)A(NC * 4)};
  float*  ktmp  = (float*)A(NC * 4);
  float*  res_u = (float*)A(NC * 4);
  float*  res_i = (float*)A(NC * 4);
  float*  a_src = (float*)A((size_t)NN * 4);
  float*  a_dst = (float*)A((size_t)NN * 4);
  unsigned* amax = (unsigned*)A((size_t)NN * 4);
  float*  denom = (float*)A((size_t)NN * 4);
  float*  alpha = (float*)A((size_t)NEDG * 4);
  float*  eexp  = (float*)A((size_t)NEDG * 4);
  float*  kmean = (float*)A(2 * CH * 4);
  float*  attnb = (float*)A(256);
  float*  post_u = (float*)A((size_t)NN * 2 * 4);
  float*  post_i = (float*)A((size_t)NN * 2 * 4);

  const int gNC   = (iNC + 255) / 256;
  const int gWW   = (CH * CH + 255) / 256;
  const int gGEMM = NN / 16;               // 3125
  const int gN    = (NN + 255) / 256;
  const int gNw   = (NN + 7) / 8;
  const int gE    = (NEDG + 255) / 256;
  const int gEw   = (NEDG + 7) / 8;
  const int gCS   = (NN + 63) / 64;

  auto run_conv = [&](const float* hu, const float* hi, int base,
                      float* resu, float* resi) {
    const float* att_dst[3] = {(const float*)d_in[base + 2],   // rates
                               (const float*)d_in[base + 0],   // rated_by
                               (const float*)d_in[base + 1]};  // follows
    const float* att_src[3] = {(const float*)d_in[base + 5],
                               (const float*)d_in[base + 3],
                               (const float*)d_in[base + 4]};
    const float* klb  = (const float*)d_in[base + 6];
    const float* klw  = (const float*)d_in[base + 7];
    const float* pb_i = (const float*)d_in[base + 8];
    const float* pw_i = (const float*)d_in[base + 9];
    const float* pb_u = (const float*)d_in[base + 10];
    const float* pw_u = (const float*)d_in[base + 11];
    const float* qv   = (const float*)d_in[base + 12];

    // projections: h = X @ W + b   (WMMA, bf16 in / f32 out)
    k_f32_to_bf16<<<gNC, 256, 0, stream>>>(hu, xb_u, iNC);
    k_f32_to_bf16<<<gNC, 256, 0, stream>>>(hi, xb_i, iNC);
    k_wtrans<<<gWW, 256, 0, stream>>>(pw_u, wb);
    k_wmma_gemm<<<gGEMM, 256, 0, stream>>>(xb_u, wb, pb_u, hp_u, 0);
    k_wtrans<<<gWW, 256, 0, stream>>>(pw_i, wb);
    k_wmma_gemm<<<gGEMM, 256, 0, stream>>>(xb_i, wb, pb_i, hp_i, 0);

    // edge types: 0 rates u->i, 1 rated_by i->u, 2 follows u->u
    const float* srcH[3] = {hp_u, hp_i, hp_u};
    const float* dstH[3] = {hp_i, hp_u, hp_u};
    for (int et = 0; et < 3; ++et) {
      k_attdot<<<gNw, 256, 0, stream>>>(srcH[et], att_src[et], a_src, NN);
      k_attdot<<<gNw, 256, 0, stream>>>(dstH[et], att_dst[et], a_dst, NN);
      k_fill_u32<<<gN, 256, 0, stream>>>(amax, ORD_NEG_INF, NN);
      k_fill_u32<<<gN, 256, 0, stream>>>((unsigned*)denom, 0u, NN);
      k_fill_u32<<<gNC, 256, 0, stream>>>((unsigned*)outb[et], 0u, iNC);
      k_edge_max<<<gE, 256, 0, stream>>>(ei[et], a_src, a_dst, alpha, amax, NEDG);
      k_edge_exp<<<gE, 256, 0, stream>>>(ei[et], alpha, amax, eexp, denom, NEDG);
      k_edge_msg<<<gEw, 256, 0, stream>>>(ei[et], eexp, denom, srcH[et], outb[et], NEDG);
      k_relu<<<gNC, 256, 0, stream>>>(outb[et], iNC);
    }

    k_wtrans<<<gWW, 256, 0, stream>>>(klw, wb);

    // semantic attention: item (K=1, stack = {rates})
    k_fill_u32<<<1, 256, 0, stream>>>((unsigned*)kmean, 0u, 2 * CH);
    k_f32_to_bf16<<<gNC, 256, 0, stream>>>(outb[0], xb_u, iNC);
    k_wmma_gemm<<<gGEMM, 256, 0, stream>>>(xb_u, wb, klb, ktmp, 2);
    k_colsum<<<gCS, 256, 0, stream>>>(ktmp, kmean);
    k_attn<<<1, 32, 0, stream>>>(qv, kmean, attnb, 1);
    k_combine<<<gNC, 256, 0, stream>>>(outb[0], outb[0], attnb, 1, resi, iNC);

    // semantic attention: user (K=2, stack = {rated_by, follows})
    k_fill_u32<<<1, 256, 0, stream>>>((unsigned*)kmean, 0u, 2 * CH);
    k_f32_to_bf16<<<gNC, 256, 0, stream>>>(outb[1], xb_u, iNC);
    k_wmma_gemm<<<gGEMM, 256, 0, stream>>>(xb_u, wb, klb, ktmp, 2);
    k_colsum<<<gCS, 256, 0, stream>>>(ktmp, kmean);
    k_f32_to_bf16<<<gNC, 256, 0, stream>>>(outb[2], xb_u, iNC);
    k_wmma_gemm<<<gGEMM, 256, 0, stream>>>(xb_u, wb, klb, ktmp, 2);
    k_colsum<<<gCS, 256, 0, stream>>>(ktmp, kmean + CH);
    k_attn<<<1, 32, 0, stream>>>(qv, kmean, attnb, 2);
    k_combine<<<gNC, 256, 0, stream>>>(outb[1], outb[2], attnb, 2, resu, iNC);
  };

  // layer 1 (inputs raw features) -> res (includes top-level relu)
  run_conv(x_user, x_item, C1, res_u, res_i);
  // layer 2 (inputs = relu'd layer-1 output); hp_* are only used as proj temps
  // before the final combines, so they safely receive the layer-2 result.
  run_conv(res_u, res_i, C2, hp_u, hp_i);

  // post dense: (NN x 128) @ (128 x 2) + b
  k_post<<<gNw, 256, 0, stream>>>(hp_u, post_w, post_b, post_u, NN);
  k_post<<<gNw, 256, 0, stream>>>(hp_i, post_w, post_b, post_i, NN);

  // scoring: rates (u->i), rated_by (i->u), follows (u->u)
  float* outp = (float*)d_out;
  const int gL = (NLBL + 255) / 256;
  k_score<<<gL, 256, 0, stream>>>(lbl[0], post_u, post_i, rel_emb, 0, outp + 0 * NLBL, NLBL);
  k_score<<<gL, 256, 0, stream>>>(lbl[1], post_i, post_u, rel_emb, 1, outp + 1 * NLBL, NLBL);
  k_score<<<gL, 256, 0, stream>>>(lbl[2], post_u, post_u, rel_emb, 2, outp + 2 * NLBL, NLBL);
}
